// MLP_with_sentiment_etc_6193342841637
// MI455X (gfx1250) — matrix operand order repaired and checked by reading, not compiled, not run on hardware
//
#include <hip/hip_runtime.h>
#include <hip/hip_bf16.h>

// ---------------------------------------------------------------------------
// CDNA5 (gfx1250) transformer forward. wave32. WMMA bf16 16x16x32 everywhere.
// All WMMA fragments but Q are loaded with ds_load_b128 via layout-aware LDS tiling
// (A tiles row-major [m][k], B/V tiles transposed [n][k]); GEMM is double-
// buffered with register staging.
// ---------------------------------------------------------------------------

typedef __attribute__((ext_vector_type(16))) __bf16 bfx16;
typedef __attribute__((ext_vector_type(8)))  __bf16 bfx8;
typedef __attribute__((ext_vector_type(8)))  float  f32x8;
typedef __attribute__((ext_vector_type(8)))  unsigned short u16x8;

#define BB   4
#define SS   1024
#define DD   2048
#define HH   16
#define HD   128
#define DFF  8192
#define VV   32000
#define NROWS (BB*SS)          // 4096
#define NT_HEAD (VV/128)       // 250

__device__ inline f32x8 zero8() {
  f32x8 z;
#pragma unroll
  for (int i = 0; i < 8; ++i) z[i] = 0.f;
  return z;
}

__device__ inline bfx16 cat8(bfx8 lo, bfx8 hi) {
  return __builtin_shufflevector(lo, hi, 0, 1, 2, 3, 4, 5, 6, 7,
                                 8, 9, 10, 11, 12, 13, 14, 15);
}

__device__ inline f32x8 wmma_bf16(bfx16 a, bfx16 b, f32x8 c) {
  return __builtin_amdgcn_wmma_f32_16x16x32_bf16(false, a, false, b, (short)0, c, false, false);
}

// Fragment contiguity (per ISA layout tables):
//  A (16x32, 16-bit): lane half h -> elems 0..7 = K[h*8 .. h*8+7],
//                                    elems 8..15 = K[16+h*8 .. 16+h*8+7]
//  B (32x16, 16-bit): same as A; this code uses elems 0..15 = K[h*16 .. h*16+15] for A and B
// => A frag = two b128 from a row-major [row][k] tile,
//    B frag = two b128 from a transposed [n][k] tile.

// ---------------------------------------------------------------------------
// Embedding gather
// ---------------------------------------------------------------------------
__global__ __launch_bounds__(256) void embed_kernel(const int* __restrict__ ids,
                                                    const float* __restrict__ emb,
                                                    float* __restrict__ x) {
  const int row = blockIdx.x;
  const int id = ids[row];
  const float4* src = (const float4*)(emb + (size_t)id * DD);
  float4* dst = (float4*)(x + (size_t)row * DD);
  for (int i = threadIdx.x; i < DD / 4; i += 256) dst[i] = src[i];
}

// ---------------------------------------------------------------------------
// LayerNorm: fp32 in -> bf16 out (+ optional fp32 copy)
// ---------------------------------------------------------------------------
__global__ __launch_bounds__(256) void layernorm_kernel(const float* __restrict__ x,
                                                        const float* __restrict__ g,
                                                        const float* __restrict__ bb,
                                                        __bf16* __restrict__ obf,
                                                        float* __restrict__ of) {
  __shared__ float red[256];
  const int row = blockIdx.x, tid = threadIdx.x;
  const float* xr = x + (size_t)row * DD;
  float v[8];
  float s = 0.f;
#pragma unroll
  for (int i = 0; i < 8; ++i) { v[i] = xr[tid + i * 256]; s += v[i]; }
  red[tid] = s; __syncthreads();
  for (int o2 = 128; o2 > 0; o2 >>= 1) { if (tid < o2) red[tid] += red[tid + o2]; __syncthreads(); }
  const float mean = red[0] * (1.f / DD);
  __syncthreads();
  float s2 = 0.f;
#pragma unroll
  for (int i = 0; i < 8; ++i) { float d = v[i] - mean; s2 += d * d; }
  red[tid] = s2; __syncthreads();
  for (int o2 = 128; o2 > 0; o2 >>= 1) { if (tid < o2) red[tid] += red[tid + o2]; __syncthreads(); }
  const float rstd = rsqrtf(red[0] * (1.f / DD) + 1e-5f);
#pragma unroll
  for (int i = 0; i < 8; ++i) {
    const int c = tid + i * 256;
    const float y = (v[i] - mean) * rstd * g[c] + bb[c];
    obf[(size_t)row * DD + c] = (__bf16)y;
    if (of) of[(size_t)row * DD + c] = y;
  }
}

// ---------------------------------------------------------------------------
// GEMM: C[M,N] = A[M,K](bf16) * B[K,N](f32->bf16 on the fly)
// MODE 0: store bf16 | MODE 1: += f32 (residual) | MODE 2: gelu->bf16
// MODE 3: LM-head logsumexp partials
// Block 256 = 8 waves (4x2). Tile 128x128, K-step 32, double-buffered LDS.
// ---------------------------------------------------------------------------
template <int MODE>
__global__ __launch_bounds__(256) void gemm_bf16_kernel(
    const __bf16* __restrict__ A, const float* __restrict__ Bw,
    float* __restrict__ Cf, __bf16* __restrict__ Cb, float2* __restrict__ part,
    int M, int N, int K, int NT) {
  __shared__ __bf16 As[2][128][48];   // [buf][m][k], row stride 96B
  __shared__ __bf16 Bs[2][128][48];   // [buf][n][k] (transposed), row stride 96B
  __shared__ float pm2[2][128];
  __shared__ float ps2[2][128];

  const int tid = threadIdx.x, lane = tid & 31, wid = tid >> 5;
  const int wm = wid & 3, wn = wid >> 2;
  const int col = lane & 15, half = lane >> 4;
  const int Mblk = blockIdx.y * 128, Nblk = blockIdx.x * 128;

  f32x8 acc[2][4];
#pragma unroll
  for (int ms = 0; ms < 2; ++ms)
#pragma unroll
    for (int ns = 0; ns < 4; ++ns) acc[ms][ns] = zero8();

  const int arow = tid >> 1, aseg = (tid & 1) * 16;   // A: 16 bf16 per thread
  const int bn = tid & 127, bkg = (tid >> 7) * 16;    // B: 16 K-strided f32 per thread

  u16x8 a0, a1;
  float bst[16];

  auto stage = [&](int kt) {   // global -> registers (issued early)
    const __bf16* ag = A + (size_t)(Mblk + arow) * K + kt + aseg;
    a0 = *(const u16x8*)ag;
    a1 = *(const u16x8*)(ag + 8);
    const float* bg = Bw + (size_t)(kt + bkg) * N + Nblk + bn;
#pragma unroll
    for (int i = 0; i < 16; ++i) bst[i] = bg[(size_t)i * N];
  };
  auto commit = [&](int buf) { // registers -> LDS (vector stores)
    *(u16x8*)&As[buf][arow][aseg] = a0;
    *(u16x8*)&As[buf][arow][aseg + 8] = a1;
    bfx8 t0, t1;
#pragma unroll
    for (int i = 0; i < 8; ++i) { t0[i] = (__bf16)bst[i]; t1[i] = (__bf16)bst[8 + i]; }
    *(bfx8*)&Bs[buf][bn][bkg] = t0;
    *(bfx8*)&Bs[buf][bn][bkg + 8] = t1;
  };

  stage(0);
  int buf = 0;
  for (int kt = 0; kt < K; kt += 32, buf ^= 1) {
    commit(buf);
    __syncthreads();
    if (kt + 32 < K) stage(kt + 32);   // overlap next tile's global loads

    bfx16 af[2], bfr[4];
#pragma unroll
    for (int ms = 0; ms < 2; ++ms) {
      const __bf16* ap = &As[buf][wm * 32 + ms * 16 + col][half * 16];
      af[ms] = cat8(*(const bfx8*)ap, *(const bfx8*)(ap + 8));
    }
#pragma unroll
    for (int ns = 0; ns < 4; ++ns) {
      const __bf16* bp = &Bs[buf][wn * 64 + ns * 16 + col][half * 16];
      bfr[ns] = cat8(*(const bfx8*)bp, *(const bfx8*)(bp + 8));
    }
#pragma unroll
    for (int ms = 0; ms < 2; ++ms)
#pragma unroll
      for (int ns = 0; ns < 4; ++ns)
        acc[ms][ns] = wmma_bf16(af[ms], bfr[ns], acc[ms][ns]);
  }

  if (MODE == 3) {
#pragma unroll
    for (int ms = 0; ms < 2; ++ms) {
#pragma unroll
      for (int r = 0; r < 8; ++r) {
        float vmax = acc[ms][0][r];
#pragma unroll
        for (int ns = 1; ns < 4; ++ns) vmax = fmaxf(vmax, acc[ms][ns][r]);
#pragma unroll
        for (int off = 8; off > 0; off >>= 1) vmax = fmaxf(vmax, __shfl_xor(vmax, off, 16));
        float se = 0.f;
#pragma unroll
        for (int ns = 0; ns < 4; ++ns) se += __expf(acc[ms][ns][r] - vmax);
#pragma unroll
        for (int off = 8; off > 0; off >>= 1) se += __shfl_xor(se, off, 16);
        if (col == 0) {
          const int rl = wm * 32 + ms * 16 + half * 8 + r;
          pm2[wn][rl] = vmax;
          ps2[wn][rl] = se;
        }
      }
    }
    __syncthreads();
    if (tid < 128) {
      const float m0 = pm2[0][tid], m1 = pm2[1][tid];
      const float nm = fmaxf(m0, m1);
      const float ssv = ps2[0][tid] * __expf(m0 - nm) + ps2[1][tid] * __expf(m1 - nm);
      part[(size_t)(Mblk + tid) * NT + blockIdx.x] = make_float2(nm, ssv);
    }
    return;
  }

#pragma unroll
  for (int ms = 0; ms < 2; ++ms) {
#pragma unroll
    for (int ns = 0; ns < 4; ++ns) {
#pragma unroll
      for (int r = 0; r < 8; ++r) {
        const int grow = Mblk + wm * 32 + ms * 16 + half * 8 + r;
        const int gcol = Nblk + wn * 64 + ns * 16 + col;
        const size_t idx = (size_t)grow * N + gcol;
        const float v2 = acc[ms][ns][r];
        if (MODE == 0) {
          Cb[idx] = (__bf16)v2;
        } else if (MODE == 1) {
          Cf[idx] += v2;
        } else {
          const float t3 = v2 * v2 * v2;
          const float gl = 0.5f * v2 * (1.f + tanhf(0.7978845608028654f * (v2 + 0.044715f * t3)));
          Cb[idx] = (__bf16)gl;
        }
      }
    }
  }
}

// ---------------------------------------------------------------------------
// Flash attention. Block = 128 threads = 4 waves; wave handles 16 query rows.
// Grid: (S/64, H, B). Online softmax over 32-key chunks; WMMA for QK^T and PV.
// V tile is stored TRANSPOSED in LDS ([dim][key]) so V fragments are two
// ds_load_b128 like every other fragment.
// ---------------------------------------------------------------------------
__global__ __launch_bounds__(128) void flash_attn_kernel(const __bf16* __restrict__ qkv,
                                                         const int* __restrict__ amask,
                                                         __bf16* __restrict__ o) {
  const int qt = blockIdx.x, hh = blockIdx.y, b = blockIdx.z;
  const int tid = threadIdx.x, lane = tid & 31, wid = tid >> 5;
  const int col = lane & 15, half = lane >> 4;
  const int qbase = qt * 64 + wid * 16;

  __shared__ __bf16 Kt[32][136];     // [key][dim]
  __shared__ __bf16 VtT[128][48];    // [dim][key]  (transposed)
  __shared__ __bf16 Pst[4][16][48];  // per-wave P staging [row][key]

  // Q fragments: two contiguous b128 global loads per 32-K chunk
  bfx16 qf[4];
  {
    const size_t qoff = ((size_t)(b * SS + qbase + col)) * (3 * DD) + hh * HD;
#pragma unroll
    for (int c = 0; c < 4; ++c) {
      const __bf16* qp = qkv + qoff + c * 32 + half * 16;
      qf[c] = cat8(*(const bfx8*)qp, *(const bfx8*)(qp + 8));
    }
  }

  f32x8 accO[8];
#pragma unroll
  for (int n = 0; n < 8; ++n) accO[n] = zero8();
  float mrow[8], lrow[8];
#pragma unroll
  for (int r = 0; r < 8; ++r) { mrow[r] = -1e30f; lrow[r] = 0.f; }

  const int nch = qt * 2 + 2;          // keys [0, qt*64+64)
  const int krow = tid >> 2, kseg = (tid & 3) * 32;
  const float scale = 0.08838834764831845f;   // 1/sqrt(128)

  for (int ch = 0; ch < nch; ++ch) {
    const int kbase = ch * 32;
    __syncthreads();
    {
      // K: coalesced vector loads, row-major [key][dim]
      const size_t src = ((size_t)(b * SS + kbase + krow)) * (3 * DD) + hh * HD + kseg;
#pragma unroll
      for (int i = 0; i < 4; ++i)
        *(u16x8*)&Kt[krow][kseg + i * 8] = *(const u16x8*)&qkv[src + DD + i * 8];
      // V transposed: each thread owns one dim, gathers its 32 keys (L2-hot),
      // writes one VtT row with vector stores.
      const size_t vsrc = ((size_t)(b * SS + kbase)) * (3 * DD) + 2 * DD + hh * HD + tid;
#pragma unroll
      for (int g2 = 0; g2 < 4; ++g2) {
        bfx8 rv;
#pragma unroll
        for (int k2 = 0; k2 < 8; ++k2)
          rv[k2] = qkv[vsrc + (size_t)(g2 * 8 + k2) * (3 * DD)];
        *(bfx8*)&VtT[tid][g2 * 8] = rv;
      }
    }
    __syncthreads();

    // scores: QK^T. K-frag = two b128 from Kt[key][dim] (contiguous in dim)
    f32x8 sc[2];
    sc[0] = zero8(); sc[1] = zero8();
#pragma unroll
    for (int t = 0; t < 2; ++t) {
#pragma unroll
      for (int c = 0; c < 4; ++c) {
        const __bf16* kp = &Kt[t * 16 + col][c * 32 + half * 16];
        const bfx16 bfr = cat8(*(const bfx8*)kp, *(const bfx8*)(kp + 8));
        sc[t] = wmma_bf16(qf[c], bfr, sc[t]);
      }
    }

    float al[8];
#pragma unroll
    for (int r = 0; r < 8; ++r) {
      const int qrow = qbase + half * 8 + r;
      float v0 = sc[0][r] * scale, v1 = sc[1][r] * scale;
      const int k0i = kbase + col, k1i = kbase + 16 + col;
      if (k0i > qrow || amask[b * SS + k0i] == 0) v0 = -1e30f;
      if (k1i > qrow || amask[b * SS + k1i] == 0) v1 = -1e30f;
      float vm = fmaxf(v0, v1);
#pragma unroll
      for (int off = 8; off > 0; off >>= 1) vm = fmaxf(vm, __shfl_xor(vm, off, 16));
      const float nm = fmaxf(mrow[r], vm);
      const float a = __expf(mrow[r] - nm);
      const float p0 = (v0 < -1e29f) ? 0.f : __expf(v0 - nm);
      const float p1 = (v1 < -1e29f) ? 0.f : __expf(v1 - nm);
      Pst[wid][half * 8 + r][col] = (__bf16)p0;
      Pst[wid][half * 8 + r][16 + col] = (__bf16)p1;
      float rs = p0 + p1;
#pragma unroll
      for (int off = 8; off > 0; off >>= 1) rs += __shfl_xor(rs, off, 16);
      lrow[r] = lrow[r] * a + rs;
      mrow[r] = nm;
      al[r] = a;
    }
#pragma unroll
    for (int n = 0; n < 8; ++n)
#pragma unroll
      for (int r = 0; r < 8; ++r) accO[n][r] *= al[r];

    __syncthreads();

    // P frag: two b128 from Pst (row-major [row][key])
    bfx16 pf;
    {
      const __bf16* pp = &Pst[wid][col][half * 16];
      pf = cat8(*(const bfx8*)pp, *(const bfx8*)(pp + 8));
    }
    // O += P x V : V-frag = two b128 from VtT[dim][key] (contiguous in key)
#pragma unroll
    for (int n = 0; n < 8; ++n) {
      const __bf16* vp = &VtT[n * 16 + col][half * 16];
      const bfx16 bfr = cat8(*(const bfx8*)vp, *(const bfx8*)(vp + 8));
      accO[n] = wmma_bf16(pf, bfr, accO[n]);
    }
  }

#pragma unroll
  for (int r = 0; r < 8; ++r) {
    const float inv = 1.f / fmaxf(lrow[r], 1e-30f);
    const int q = qbase + half * 8 + r;
    const size_t base = ((size_t)(b * SS + q)) * DD + hh * HD;
#pragma unroll
    for (int n = 0; n < 8; ++n)
      o[base + n * 16 + col] = (__bf16)(accO[n][r] * inv);
  }
}

// ---------------------------------------------------------------------------
// logsumexp partial reduction over the 250 N-tiles -> logZ[row]
// ---------------------------------------------------------------------------
__global__ __launch_bounds__(256) void lse_reduce_kernel(const float2* __restrict__ part,
                                                         float* __restrict__ logZ) {
  __shared__ float rm[256], rs[256];
  const int row = blockIdx.x, tid = threadIdx.x;
  float m = -1e30f, s = 0.f;
  for (int i = tid; i < NT_HEAD; i += 256) {
    const float2 p = part[(size_t)row * NT_HEAD + i];
    const float nm = fmaxf(m, p.x);
    s = s * __expf(m - nm) + p.y * __expf(p.x - nm);
    m = nm;
  }
  rm[tid] = m; rs[tid] = s; __syncthreads();
  for (int o2 = 128; o2 > 0; o2 >>= 1) {
    if (tid < o2) {
      const float m2 = rm[tid + o2], s2 = rs[tid + o2];
      const float nm = fmaxf(rm[tid], m2);
      rs[tid] = rs[tid] * __expf(rm[tid] - nm) + s2 * __expf(m2 - nm);
      rm[tid] = nm;
    }
    __syncthreads();
  }
  if (tid == 0) logZ[row] = rm[0] + __logf(rs[0]);
}

// ---------------------------------------------------------------------------
// Per-row own-token logit: tok[row] = h[row,:] . head_w[:, ids[row]]
// ---------------------------------------------------------------------------
__global__ __launch_bounds__(256) void tok_logit_kernel(const float* __restrict__ hf,
                                                        const float* __restrict__ head_w,
                                                        const int* __restrict__ ids,
                                                        float* __restrict__ tok) {
  __shared__ float red[256];
  const int row = blockIdx.x, tid = threadIdx.x;
  const int t = ids[row];
  float s = 0.f;
  for (int k = tid; k < DD; k += 256)
    s += hf[(size_t)row * DD + k] * head_w[(size_t)k * VV + t];
  red[tid] = s; __syncthreads();
  for (int o2 = 128; o2 > 0; o2 >>= 1) { if (tid < o2) red[tid] += red[tid + o2]; __syncthreads(); }
  if (tid == 0) tok[row] = red[0];
}

// ---------------------------------------------------------------------------
// Mean-pool final hidden over sequence
// ---------------------------------------------------------------------------
__global__ __launch_bounds__(256) void pool_kernel(const float* __restrict__ hf,
                                                   float* __restrict__ pooled) {
  const int idx = blockIdx.x * 256 + threadIdx.x;   // B*D = 8192
  const int b = idx >> 11, d = idx & (DD - 1);
  float s = 0.f;
  for (int si = 0; si < SS; ++si) s += hf[((size_t)(b * SS + si)) * DD + d];
  pooled[idx] = s * (1.f / SS);
}

// ---------------------------------------------------------------------------
// Reducer + surprisal + classifier -> out[4,3]
// ---------------------------------------------------------------------------
__global__ __launch_bounds__(256) void final_kernel(
    const float* __restrict__ pooled, const float* __restrict__ sentiment,
    const float* __restrict__ perplexity, const float* __restrict__ logZ,
    const float* __restrict__ tok, const int* __restrict__ amask,
    const float* __restrict__ red_w, const float* __restrict__ red_b,
    const float* __restrict__ cls_w, const float* __restrict__ cls_b,
    float* __restrict__ out) {
  __shared__ float surp[BB], cnt[BB], rr[BB][100];
  if (threadIdx.x < BB) { surp[threadIdx.x] = 0.f; cnt[threadIdx.x] = 0.f; }
  __syncthreads();
  {
    const int b = threadIdx.x >> 6;
    const int so = (threadIdx.x & 63) * 16;
    float ps = 0.f, pc = 0.f;
    for (int i = 0; i < 16; ++i) {
      const int srow = b * SS + so + i;
      const float am = (float)amask[srow];
      ps += (logZ[srow] - tok[srow]) * am;
      pc += am;
    }
    atomicAdd(&surp[b], ps);
    atomicAdd(&cnt[b], pc);
  }
  __syncthreads();
  for (int t = threadIdx.x; t < BB * 100; t += 256) {
    const int b = t / 100, j = t % 100;
    float s = red_b[j];
    for (int k = 0; k < DD; ++k) s += pooled[b * DD + k] * red_w[k * 100 + j];
    rr[b][j] = s > 0.f ? s : 0.01f * s;   // leaky_relu
  }
  __syncthreads();
  if (threadIdx.x < BB * 3) {
    const int b = threadIdx.x / 3, lab = threadIdx.x % 3;
    float s = cls_b[lab];
    for (int j = 0; j < 100; ++j) s += rr[b][j] * cls_w[j * 3 + lab];
    for (int j = 0; j < 3; ++j) s += sentiment[b * 3 + j] * cls_w[(100 + j) * 3 + lab];
    s += perplexity[b] * cls_w[103 * 3 + lab];
    const float msurp = (surp[b] / fmaxf(cnt[b], 1e-30f)) * 1.4426950408889634f; // /ln2
    s += msurp * cls_w[104 * 3 + lab];
    out[b * 3 + lab] = s;
  }
}

// ---------------------------------------------------------------------------
// Launch
// ---------------------------------------------------------------------------
extern "C" void kernel_launch(void* const* d_in, const int* in_sizes, int n_in,
                              void* d_out, int out_size, void* d_ws, size_t ws_size,
                              hipStream_t stream) {
  const int* ids = (const int*)d_in[0];
  const int* am = (const int*)d_in[1];
  const float* sentiment = (const float*)d_in[2];
  const float* perplexity = (const float*)d_in[3];
  const float* emb = (const float*)d_in[4];
  const float* ln1_g = (const float*)d_in[5];
  const float* ln1_b = (const float*)d_in[6];
  const float* qkv_w = (const float*)d_in[7];
  const float* wo_w = (const float*)d_in[8];
  const float* ln2_g = (const float*)d_in[9];
  const float* ln2_b = (const float*)d_in[10];
  const float* w1 = (const float*)d_in[11];
  const float* w2 = (const float*)d_in[12];
  const float* lnf_g = (const float*)d_in[13];
  const float* lnf_b = (const float*)d_in[14];
  const float* head_w = (const float*)d_in[15];
  const float* red_w = (const float*)d_in[16];
  const float* red_b = (const float*)d_in[17];
  const float* cls_w = (const float*)d_in[18];
  const float* cls_b = (const float*)d_in[19];
  float* outp = (float*)d_out;

  char* wsb = (char*)d_ws;
  size_t off = 0;
  auto alloc = [&](size_t bytes) -> void* {
    void* p = wsb + off;
    off = (off + bytes + 255) & ~(size_t)255;
    return p;
  };
  float* x = (float*)alloc((size_t)NROWS * DD * 4);
  __bf16* hbf = (__bf16*)alloc((size_t)NROWS * DD * 2);
  float* hfin = (float*)alloc((size_t)NROWS * DD * 4);
  __bf16* qkvb = (__bf16*)alloc((size_t)NROWS * 3 * DD * 2);
  __bf16* obf = (__bf16*)alloc((size_t)NROWS * DD * 2);
  __bf16* ffb = (__bf16*)alloc((size_t)NROWS * DFF * 2);
  float2* part = (float2*)alloc((size_t)NROWS * NT_HEAD * 8);
  float* logZ = (float*)alloc(NROWS * 4);
  float* tokl = (float*)alloc(NROWS * 4);
  float* pooled = (float*)alloc(BB * DD * 4);

  embed_kernel<<<NROWS, 256, 0, stream>>>(ids, emb, x);

  for (int l = 0; l < 2; ++l) {
    layernorm_kernel<<<NROWS, 256, 0, stream>>>(x, ln1_g + l * DD, ln1_b + l * DD, hbf, nullptr);
    gemm_bf16_kernel<0><<<dim3(3 * DD / 128, NROWS / 128), 256, 0, stream>>>(
        hbf, qkv_w + (size_t)l * DD * 3 * DD, nullptr, qkvb, nullptr, NROWS, 3 * DD, DD, 0);
    flash_attn_kernel<<<dim3(SS / 64, HH, BB), 128, 0, stream>>>(qkvb, am, obf);
    gemm_bf16_kernel<1><<<dim3(DD / 128, NROWS / 128), 256, 0, stream>>>(
        obf, wo_w + (size_t)l * DD * DD, x, nullptr, nullptr, NROWS, DD, DD, 0);
    layernorm_kernel<<<NROWS, 256, 0, stream>>>(x, ln2_g + l * DD, ln2_b + l * DD, hbf, nullptr);
    gemm_bf16_kernel<2><<<dim3(DFF / 128, NROWS / 128), 256, 0, stream>>>(
        hbf, w1 + (size_t)l * DD * DFF, nullptr, ffb, nullptr, NROWS, DFF, DD, 0);
    gemm_bf16_kernel<1><<<dim3(DD / 128, NROWS / 128), 256, 0, stream>>>(
        ffb, w2 + (size_t)l * DFF * DD, x, nullptr, nullptr, NROWS, DD, DFF, 0);
  }

  layernorm_kernel<<<NROWS, 256, 0, stream>>>(x, lnf_g, lnf_b, hbf, hfin);
  gemm_bf16_kernel<3><<<dim3(NT_HEAD, NROWS / 128), 256, 0, stream>>>(
      hbf, head_w, nullptr, nullptr, part, NROWS, VV, DD, NT_HEAD);
  lse_reduce_kernel<<<NROWS, 256, 0, stream>>>(part, logZ);
  tok_logit_kernel<<<NROWS, 256, 0, stream>>>(hfin, head_w, ids, tokl);
  pool_kernel<<<BB * DD / 256, 256, 0, stream>>>(hfin, pooled);
  final_kernel<<<1, 256, 0, stream>>>(pooled, sentiment, perplexity, logZ, tokl, am,
                                      red_w, red_b, cls_w, cls_b, outp);
}
